// FTDEA_78950088835244
// MI455X (gfx1250) — matrix operand order repaired and checked by reading, not compiled, not run on hardware
//
#include <hip/hip_runtime.h>
#include <cstdint>
#include <cstddef>

// ---------------------------------------------------------------------------
// FTDEA forward for MI455X (gfx1250, wave32).
// Dense GEMMs (highway gates, tc1, per-relation, and the r2e aggregation
// reformulated as W @ er) run on v_wmma_f32_16x16x32_bf16 with f32 accum.
// GEMM inner loop is branch-free: OOB rows/cols are clamped (their outputs
// are never stored), only the K-tail masks the B operand via cndmask selects.
// f32->bf16 uses native casts so the compiler emits hardware cvt ops.
// Edge work is atomic scatter/gather sized to stay L2-resident (192 MB).
// ---------------------------------------------------------------------------

typedef __attribute__((ext_vector_type(16))) __bf16 v16bf;
typedef __attribute__((ext_vector_type(8)))  float  v8f;

namespace {
constexpr int NN   = 50000;   // nodes
constexpr int EHD  = 300;     // e_hidden
constexpr int TDIM = 100;     // t_hidden
constexpr int RHD  = 100;     // r_hidden
constexpr int NREL = 300;     // relations
constexpr int NE   = 250000;  // typed edges
constexpr int NEA  = 500000;  // all edges
constexpr int EP_BIAS = 0, EP_HIGHWAY = 1, EP_ADDRELU = 2;
constexpr int MT   = 4;       // 16-row subtiles per wave (64 rows / wave)
}

// order-preserving float <-> uint encoding for atomicMax-based segment max
__device__ __forceinline__ unsigned int fenc(float f) {
  union { float f; unsigned int u; } c; c.f = f;
  return (c.u & 0x80000000u) ? ~c.u : (c.u | 0x80000000u);
}
__device__ __forceinline__ float fdec(unsigned int u) {
  union { float f; unsigned int u; } c;
  c.u = (u & 0x80000000u) ? (u & 0x7FFFFFFFu) : ~u;
  return c.f;
}

// ---------------------------------------------------------------------------
// WMMA GEMM: C[M,N] = epilogue(A[M,K] @ B[K,N] + bias)
// One wave computes a 64x16 C slab (4 stacked 16x16 WMMA tiles sharing one
// B fragment). Fragment layouts per CDNA5 ISA 7.12.2 (wave32).
// ---------------------------------------------------------------------------
__global__ __launch_bounds__(256)
void k_gemm_bf16(const float* A, int lda, const float* __restrict__ B, int ldb,
                 const float* __restrict__ bias, float* C, int ldc,
                 const float* P1, const float* P2,
                 int M, int N, int K, int mode)
{
  const int wave = (int)((blockIdx.x * blockDim.x + threadIdx.x) >> 5);
  const int lane = (int)(threadIdx.x & 31u);
  const int tn   = (N + 15) >> 4;
  const int tmg  = (M + 16 * MT - 1) / (16 * MT);
  if (wave >= tmg * tn) return;           // wave-uniform: EXEC all-1s at WMMA
  const int m0 = (wave / tn) * 16 * MT;
  const int n0 = (wave % tn) * 16;
  const int l15 = lane & 15;
  const int kh  = lane >> 4;              // lane half (0: lanes 0-15, 1: 16-31)

  // A: lane carries row m (sub-tile t -> m0 + 16t + l15), clamped in-bounds.
  const float* Aptr[MT];
#pragma unroll
  for (int t = 0; t < MT; ++t) {
    int r = m0 + 16 * t + l15;
    if (r > M - 1) r = M - 1;             // clamped rows: outputs never stored
    Aptr[t] = A + (size_t)r * lda;
  }
  // B: lane carries column n, clamped in-bounds.
  const int bn = n0 + l15;
  const float* Bcol = B + (bn > N - 1 ? N - 1 : bn);

  v8f acc[MT];
#pragma unroll
  for (int t = 0; t < MT; ++t) acc[t] = (v8f){};

  int k0 = 0;
  // ---- main loop: fully unconditional loads, 4 WMMAs per B fragment ----
  for (; k0 + 32 <= K; k0 += 32) {
    v16bf fb;
#pragma unroll
    for (int v = 0; v < 8; ++v) {
      // B 32x16 bf16 layout: same K order as A: v<4 -> K=2v, v>=4 -> K=16+2(v-4); +8 hi half
      const int kb = k0 + ((v < 4) ? (2 * v) : (16 + 2 * (v - 4))) + kh * 8;
      fb[2 * v]     = (__bf16)Bcol[(size_t)kb * ldb];        // native cvt
      fb[2 * v + 1] = (__bf16)Bcol[(size_t)(kb + 1) * ldb];
    }
#pragma unroll
    for (int t = 0; t < MT; ++t) {
      v16bf fa;
#pragma unroll
      for (int v = 0; v < 8; ++v) {
        // A 16x32 bf16 layout: VGPR v<4 -> K=2v, v>=4 -> K=16+2(v-4); +8 hi half
        const int ka = k0 + ((v < 4) ? (2 * v) : (16 + 2 * (v - 4))) + kh * 8;
        const float2 av = *(const float2*)(Aptr[t] + ka);    // merges to b128
        fa[2 * v]     = (__bf16)av.x;
        fa[2 * v + 1] = (__bf16)av.y;
      }
      acc[t] = __builtin_amdgcn_wmma_f32_16x16x32_bf16(false, fa, false, fb,
                                                       (short)0, acc[t], false, false);
    }
  }
  // ---- K tail: branch-free (index clamp on A, value mask on B) ----
  if (k0 < K) {
    v16bf fb;
#pragma unroll
    for (int v = 0; v < 8; ++v) {
      const int kb = k0 + ((v < 4) ? (2 * v) : (16 + 2 * (v - 4))) + kh * 8;
      const bool o0 = kb < K, o1 = kb + 1 < K;
      float b0 = Bcol[(size_t)(o0 ? kb : 0) * ldb];
      float b1 = Bcol[(size_t)(o1 ? kb + 1 : 0) * ldb];
      b0 = o0 ? b0 : 0.0f;
      b1 = o1 ? b1 : 0.0f;                // zero B => k>=K contributes exactly 0
      fb[2 * v]     = (__bf16)b0;
      fb[2 * v + 1] = (__bf16)b1;
    }
#pragma unroll
    for (int t = 0; t < MT; ++t) {
      v16bf fa;
#pragma unroll
      for (int v = 0; v < 8; ++v) {
        const int ka = k0 + ((v < 4) ? (2 * v) : (16 + 2 * (v - 4))) + kh * 8;
        const float2 av = *(const float2*)(Aptr[t] + (ka + 1 < K ? ka : 0));
        fa[2 * v]     = (__bf16)av.x;     // finite garbage * 0 == 0
        fa[2 * v + 1] = (__bf16)av.y;
      }
      acc[t] = __builtin_amdgcn_wmma_f32_16x16x32_bf16(false, fa, false, fb,
                                                       (short)0, acc[t], false, false);
    }
  }

  // ---- epilogue ----
  if (bn >= N) return;                    // divergence only after last WMMA
  const float bv = bias ? bias[bn] : 0.0f;
#pragma unroll
  for (int t = 0; t < MT; ++t) {
#pragma unroll
    for (int r = 0; r < 8; ++r) {
      // C/D layout: lanes 0-15 -> M=r, lanes 16-31 -> M=8+r, N = lane&15
      const int cm = m0 + 16 * t + kh * 8 + r;
      if (cm >= M) continue;
      const size_t idx = (size_t)cm * ldc + bn;
      float v = acc[t][r] + bv;
      if (mode == EP_HIGHWAY) {           // x' = g*relu(gcn) + (1-g)*x
        const float gate = 1.0f / (1.0f + __expf(-v));
        float g = P1[idx]; g = g > 0.0f ? g : 0.0f;
        v = gate * g + (1.0f - gate) * P2[idx];
      } else if (mode == EP_ADDRELU) {    // x' = x + relu(W@er)
        v = P2[idx] + (v > 0.0f ? v : 0.0f);
      }
      C[idx] = v;
    }
  }
}

// ---------------------------------------------------------------------------
// Utility / edge kernels
// ---------------------------------------------------------------------------
__global__ void k_zero(float* p, long long n) {
  long long g = (long long)blockIdx.x * blockDim.x + threadIdx.x;
  if (g < n) p[g] = 0.0f;
}

__global__ void k_deg(const int* ii, float* deg, int E) {
  int g = blockIdx.x * blockDim.x + threadIdx.x;
  if (g < E) atomicAdd(&deg[ii[g]], 1.0f);
}

__global__ void k_dinv(const float* deg, float* dinv, int n) {
  int g = blockIdx.x * blockDim.x + threadIdx.x;
  if (g < n) { float d = deg[g]; dinv[g] = d > 0.0f ? rsqrtf(fmaxf(d, 1.0f)) : 0.0f; }
}

__global__ __launch_bounds__(256)
void k_gcn_scatter(const float* x, const int* jj, const int* ii,
                   const float* dinv, float* outp, long long total)
{
  long long g = (long long)blockIdx.x * blockDim.x + threadIdx.x;
  if (g >= total) return;
  const int e = (int)(g / EHD), c = (int)(g % EHD);
  const int j = jj[e], i = ii[e];
  const float w = dinv[j] * dinv[i];
  atomicAdd(&outp[(size_t)i * EHD + c], w * x[(size_t)j * EHD + c]);
}

__global__ void k_relcnt(const int* rel, float* cnt, int E) {
  int g = blockIdx.x * blockDim.x + threadIdx.x;
  if (g < E) atomicAdd(&cnt[rel[g]], 1.0f);
}

__global__ __launch_bounds__(256)
void k_relsum(const int* h, const int* t, const int* rel, const float* s,
              float* mh, float* mt, long long total)
{
  long long g = (long long)blockIdx.x * blockDim.x + threadIdx.x;
  if (g >= total) return;
  const int e = (int)(g / TDIM), c = (int)(g % TDIM);
  const int r = rel[e];
  atomicAdd(&mh[(size_t)r * TDIM + c], s[(size_t)h[e] * TDIM + c]);
  atomicAdd(&mt[(size_t)r * TDIM + c], s[(size_t)t[e] * TDIM + c]);
}

__global__ void k_reldiv(const float* cnt, float* mh, float* mt, int total) {
  int g = blockIdx.x * blockDim.x + threadIdx.x;
  if (g >= total) return;
  const float c = fmaxf(cnt[g / TDIM], 1.0f);
  mh[g] /= c; mt[g] /= c;
}

__global__ void k_build_rin(const float* mh, const float* mt, float* rin, int total) {
  int g = blockIdx.x * blockDim.x + threadIdx.x;
  if (g >= total) return;
  const int r = g / (2 * TDIM), c = g % (2 * TDIM);
  rin[g] = (c < TDIM) ? mh[r * TDIM + c] : mt[r * TDIM + (c - TDIM)];
}

// r_f = [x_type | mh | mt] per relation (x_res1==0; uniform softmax => x_type==x_res2)
__global__ void k_build_rf(const float* xres2, const float* mh, const float* mt,
                           float* rf, int total) {
  int g = blockIdx.x * blockDim.x + threadIdx.x;
  if (g >= total) return;
  const int r = g / EHD, c = g % EHD;
  rf[g] = (c < RHD) ? xres2[r * RHD + c]
        : (c < RHD + TDIM) ? mh[r * TDIM + (c - RHD)]
                           : mt[r * TDIM + (c - RHD - TDIM)];
}

// per-row dot(s): o1[w] = row_w(x) . a1 ; optional o2 with a2
__global__ __launch_bounds__(256)
void k_node_dot2(const float* x, const float* a1, const float* a2,
                 float* o1, float* o2, int n, int D)
{
  const int wv = (int)((blockIdx.x * blockDim.x + threadIdx.x) >> 5);
  const int lane = (int)(threadIdx.x & 31u);
  if (wv >= n) return;
  float s1 = 0.0f, s2 = 0.0f;
  for (int c = lane; c < D; c += 32) {
    const float xv = x[(size_t)wv * D + c];
    s1 += xv * a1[c];
    if (a2) s2 += xv * a2[c];
  }
  for (int off = 16; off > 0; off >>= 1) {
    s1 += __shfl_down(s1, off);
    s2 += __shfl_down(s2, off);
  }
  if (lane == 0) { o1[wv] = s1; if (o2) o2[wv] = s2; }
}

// l[e] = lrelu(sa[seg[e]] + sb[oth[e]]); segment max via ordered-uint atomicMax
__global__ void k_edge_max(const int* seg, const int* oth, const float* sa,
                           const float* sb, float* lbuf, unsigned int* mEnc, int E)
{
  int g = blockIdx.x * blockDim.x + threadIdx.x;
  if (g >= E) return;
  float l = sa[seg[g]] + sb[oth[g]];
  l = l > 0.0f ? l : 0.01f * l;           // leaky_relu, slope 0.01
  lbuf[g] = l;
  atomicMax(&mEnc[seg[g]], fenc(l));
}

__global__ void k_edge_expsum(const int* seg, const float* lbuf,
                              const unsigned int* mEnc, float* pbuf,
                              float* ssum, int E)
{
  int g = blockIdx.x * blockDim.x + threadIdx.x;
  if (g >= E) return;
  const int s = seg[g];
  const float p = __expf(lbuf[g] - fdec(mEnc[s]));
  pbuf[g] = p;
  atomicAdd(&ssum[s], p);
}

// attention-mass matrix: W[n, rel] += alpha  (one scalar atomic per edge)
__global__ void k_edge_scatW(const int* seg, const int* rel, const float* pbuf,
                             const float* ssum, float* W, int E)
{
  int g = blockIdx.x * blockDim.x + threadIdx.x;
  if (g >= E) return;
  const int n = seg[g];
  atomicAdd(&W[(size_t)n * NREL + rel[g]], pbuf[g] / ssum[n]);
}

// final GAT aggregation: g[i,c] += alpha * x[j,c]
__global__ __launch_bounds__(256)
void k_edge_scatG(const int* ii, const int* jj, const float* pbuf,
                  const float* ssum, const float* x, float* gout, long long total)
{
  long long g = (long long)blockIdx.x * blockDim.x + threadIdx.x;
  if (g >= total) return;
  const int e = (int)(g / EHD), c = (int)(g % EHD);
  const int i = ii[e];
  const float a = pbuf[e] / ssum[i];
  atomicAdd(&gout[(size_t)i * EHD + c], a * x[(size_t)jj[e] * EHD + c]);
}

// out[n] = [x | relu(g)] . fc_w + fc_b
__global__ __launch_bounds__(256)
void k_final(const float* x, const float* g, const float* fcw, const float* fcb,
             float* out, int n)
{
  const int wv = (int)((blockIdx.x * blockDim.x + threadIdx.x) >> 5);
  const int lane = (int)(threadIdx.x & 31u);
  if (wv >= n) return;
  float s = 0.0f;
  for (int c = lane; c < EHD; c += 32) {
    s += x[(size_t)wv * EHD + c] * fcw[c];
    float gv = g[(size_t)wv * EHD + c]; gv = gv > 0.0f ? gv : 0.0f;
    s += gv * fcw[EHD + c];
  }
  for (int off = 16; off > 0; off >>= 1) s += __shfl_down(s, off);
  if (lane == 0) out[wv] = s + fcb[0];
}

// ---------------------------------------------------------------------------
extern "C" void kernel_launch(void* const* d_in, const int* in_sizes, int n_in,
                              void* d_out, int out_size, void* d_ws, size_t ws_size,
                              hipStream_t stream)
{
  (void)in_sizes; (void)n_in; (void)out_size;

  const float* x_in  = (const float*)d_in[0];
  const int*   ei    = (const int*)d_in[1];
  const int*   rel   = (const int*)d_in[2];
  const int*   eia   = (const int*)d_in[3];
  // d_in[4] rel_size == arange(E): identity gather, unused
  const float* hw1_w = (const float*)d_in[5];  const float* hw1_b = (const float*)d_in[6];
  const float* hw2_w = (const float*)d_in[7];  const float* hw2_b = (const float*)d_in[8];
  const float* tc1_w = (const float*)d_in[9];  const float* tc1_b = (const float*)d_in[10];
  const float* sr1_w = (const float*)d_in[11]; const float* sr1_b = (const float*)d_in[12];
  // d_in[13]=a1_w, d_in[14]=a5_w: mathematically dead (softmax over per-relation
  // constant logits is exactly uniform; x_res1 is all-zeros)
  const float* wr_w  = (const float*)d_in[15]; const float* wr_b  = (const float*)d_in[16];
  const float* wr1_w = (const float*)d_in[17]; const float* wr1_b = (const float*)d_in[18];
  const float* wr2_w = (const float*)d_in[19]; const float* wr2_b = (const float*)d_in[20];
  const float* ah_w  = (const float*)d_in[21]; const float* ah1_w = (const float*)d_in[22];
  const float* at_w  = (const float*)d_in[23];
  const float* ar1_w = (const float*)d_in[24]; const float* ar2_w = (const float*)d_in[25];
  const float* ar3_w = (const float*)d_in[26];
  const float* ai_w  = (const float*)d_in[27]; const float* aj_w  = (const float*)d_in[28];
  const float* fc_w  = (const float*)d_in[29]; const float* fc_b  = (const float*)d_in[30];

  const int* hptr = ei;                 const int* tptr = ei + NE;
  const int* jall = eia;                const int* iall = eia + NEA;
  float* out = (float*)d_out;

  // --- workspace carve (fp32 elements) ---
  float* p = (float*)d_ws;
  auto take = [&](size_t n) { float* r = p; p += n; return r; };
  float* xs0  = take((size_t)NN * EHD);       // node state ping
  float* xs1  = take((size_t)NN * EHD);       // node state pong
  float* gb   = take((size_t)NN * EHD);       // gcn out / W matrix / GAT agg
  float* sb   = take((size_t)NN * TDIM);      // s = x @ tc1
  float* deg  = take(NN);  float* dinv = take(NN);
  float* xa   = take(NN);  float* xa2  = take(NN);
  float* ssum = take(NN);  unsigned int* mEnc = (unsigned int*)take(NN);
  float* lbuf = take(NEA); float* pbuf = take(NEA);
  float* cnt  = take(NREL);
  float* mh   = take((size_t)NREL * TDIM);
  float* mt   = take((size_t)NREL * TDIM);
  float* rin  = take((size_t)NREL * 2 * TDIM);
  float* xres2= take((size_t)NREL * RHD);
  float* rf   = take((size_t)NREL * EHD);
  float* er   = take((size_t)NREL * EHD);
  float* ra   = take(NREL);
  if ((size_t)((char*)p - (char*)d_ws) > ws_size) return;   // not enough scratch
  float* Wm = gb;   // gb is free during the r2e rounds; reuse as W[n,rel]

  const int TPB = 256;
  auto NB = [](long long n) { return (unsigned)((n + 255) / 256); };
  auto ZERO = [&](float* ptr, long long n) {
    k_zero<<<NB(n), TPB, 0, stream>>>(ptr, n);
  };
  auto GEMM = [&](const float* A, int lda, const float* B, int ldb,
                  const float* bias, float* C, int ldc,
                  const float* P1, const float* P2, int M, int N, int K, int mode) {
    const int tiles = ((M + 16 * MT - 1) / (16 * MT)) * ((N + 15) / 16);
    k_gemm_bf16<<<(tiles + 7) / 8, TPB, 0, stream>>>(A, lda, B, ldb, bias, C, ldc,
                                                     P1, P2, M, N, K, mode);
  };

  // --- symmetric-norm degrees (same graph for both GCN layers) ---
  ZERO(deg, NN);
  k_deg<<<NB(NEA), TPB, 0, stream>>>(iall, deg, NEA);
  k_dinv<<<NB(NN), TPB, 0, stream>>>(deg, dinv, NN);

  // --- GCN1 + highway1 (WMMA gate GEMM, fused sigmoid/relu/mix epilogue) ---
  ZERO(gb, (long long)NN * EHD);
  k_gcn_scatter<<<NB((long long)NEA * EHD), TPB, 0, stream>>>(x_in, jall, iall, dinv, gb,
                                                              (long long)NEA * EHD);
  GEMM(x_in, EHD, hw1_w, EHD, hw1_b, xs0, EHD, gb, x_in, NN, EHD, EHD, EP_HIGHWAY);

  // --- GCN2 + highway2 ---
  ZERO(gb, (long long)NN * EHD);
  k_gcn_scatter<<<NB((long long)NEA * EHD), TPB, 0, stream>>>(xs0, jall, iall, dinv, gb,
                                                              (long long)NEA * EHD);
  GEMM(xs0, EHD, hw2_w, EHD, hw2_b, xs1, EHD, gb, xs0, NN, EHD, EHD, EP_HIGHWAY);

  // --- s = x @ tc1 + b ---
  GEMM(xs1, EHD, tc1_w, TDIM, tc1_b, sb, TDIM, nullptr, nullptr, NN, TDIM, EHD, EP_BIAS);

  // --- per-relation means of s[h], s[t] ---
  ZERO(cnt, NREL);
  ZERO(mh, (long long)NREL * TDIM);
  ZERO(mt, (long long)NREL * TDIM);
  k_relcnt<<<NB(NE), TPB, 0, stream>>>(rel, cnt, NE);
  k_relsum<<<NB((long long)NE * TDIM), TPB, 0, stream>>>(hptr, tptr, rel, sb, mh, mt,
                                                         (long long)NE * TDIM);
  k_reldiv<<<NB(NREL * TDIM), TPB, 0, stream>>>(cnt, mh, mt, NREL * TDIM);

  // --- x_res2 = [mh|mt] @ sr1 + b ; r_f = [x_res2 | mh | mt] (per relation) ---
  k_build_rin<<<NB(NREL * 2 * TDIM), TPB, 0, stream>>>(mh, mt, rin, NREL * 2 * TDIM);
  GEMM(rin, 2 * TDIM, sr1_w, RHD, sr1_b, xres2, RHD, nullptr, nullptr,
       NREL, RHD, 2 * TDIM, EP_BIAS);
  k_build_rf<<<NB(NREL * EHD), TPB, 0, stream>>>(xres2, mh, mt, rf, NREL * EHD);

  // --- three r2e attention rounds (head, tail, head) ---
  const float* WRw[3] = {wr_w, wr1_w, wr2_w};
  const float* WRb[3] = {wr_b, wr1_b, wr2_b};
  const float* ANw[3] = {ah_w, at_w, ah1_w};
  const float* ARw[3] = {ar1_w, ar2_w, ar3_w};
  const int*   IDX[3] = {hptr, tptr, hptr};

  float* xc = xs1; float* xn = xs0;
  for (int rd = 0; rd < 3; ++rd) {
    // er = r_f @ wr + b  (300x300x300, WMMA)
    GEMM(rf, EHD, WRw[rd], EHD, WRb[rd], er, EHD, nullptr, nullptr,
         NREL, EHD, EHD, EP_BIAS);
    // per-relation / per-node attention scores
    k_node_dot2<<<NB(32LL * NREL), TPB, 0, stream>>>(er, ARw[rd], nullptr, ra, nullptr,
                                                     NREL, EHD);
    k_node_dot2<<<NB(32LL * NN), TPB, 0, stream>>>(xc, ANw[rd], nullptr, xa, nullptr,
                                                   NN, EHD);
    // segment softmax over target nodes
    ZERO(ssum, NN); ZERO((float*)mEnc, NN);
    k_edge_max   <<<NB(NE), TPB, 0, stream>>>(IDX[rd], rel, xa, ra, lbuf, mEnc, NE);
    k_edge_expsum<<<NB(NE), TPB, 0, stream>>>(IDX[rd], lbuf, mEnc, pbuf, ssum, NE);
    // W[n,rel] accumulation (scalar atomics), then dense WMMA: x += relu(W @ er)
    ZERO(Wm, (long long)NN * NREL);
    k_edge_scatW<<<NB(NE), TPB, 0, stream>>>(IDX[rd], rel, pbuf, ssum, Wm, NE);
    GEMM(Wm, NREL, er, EHD, nullptr, xn, EHD, nullptr, xc, NN, EHD, NREL, EP_ADDRELU);
    float* tmp = xc; xc = xn; xn = tmp;
  }

  // --- final GAT over edge_index_all ---
  k_node_dot2<<<NB(32LL * NN), TPB, 0, stream>>>(xc, ai_w, aj_w, xa, xa2, NN, EHD);
  ZERO(ssum, NN); ZERO((float*)mEnc, NN);
  k_edge_max   <<<NB(NEA), TPB, 0, stream>>>(iall, jall, xa, xa2, lbuf, mEnc, NEA);
  k_edge_expsum<<<NB(NEA), TPB, 0, stream>>>(iall, lbuf, mEnc, pbuf, ssum, NEA);
  ZERO(gb, (long long)NN * EHD);
  k_edge_scatG<<<NB((long long)NEA * EHD), TPB, 0, stream>>>(iall, jall, pbuf, ssum, xc,
                                                             gb, (long long)NEA * EHD);
  // out = [x | relu(g)] @ fc_w + fc_b
  k_final<<<NB(32LL * NN), TPB, 0, stream>>>(xc, gb, fc_w, fc_b, out, NN);
}
